// RUMLayer_77635828842710
// MI455X (gfx1250) — matrix operand order repaired and checked by reading, not compiled, not run on hardware
//
#include <hip/hip_runtime.h>
#include <math.h>

typedef _Float16 h16;
typedef __attribute__((ext_vector_type(16))) _Float16 v16h;
typedef __attribute__((ext_vector_type(8)))  float    v8f;

#define NN   10000
#define SS   4
#define LL   8
#define FF   128
#define HD   128
#define ORIG 64
#define SUBN 100
#define BB   (SS*NN)                      /* 40000 */
#define LOSS_COUNT (SS*SUBN*(LL-1)*ORIG)  /* 179200 */

// ---------------------------------------------------------------- helpers

__device__ __forceinline__ v8f wmma16(v16h a, v16h b, v8f c) {
  // v_wmma_f32_16x16x32_f16 : D = A x B + C, fp32 accumulate
  return __builtin_amdgcn_wmma_f32_16x16x32_f16(false, a, false, b, (short)0, c, false, false);
}

// A fragment (16x32 f16) from an f16 LDS tile (two contiguous 16B runs/lane
// -> 2x ds_load_b128). half i: k = kbase + (i>>3)*16 + ((ln>=16)?8:0) + (i&7)
__device__ __forceinline__ v16h load_a_half(const h16* x, int ln, int kbase) {
  union { v16h v; h16 h[16]; } u;
  const int m  = ln & 15;
  const int hi = (ln >> 4) << 3;
  const h16* p = x + m * HD + kbase + hi;
  #pragma unroll
  for (int g = 0; g < 2; ++g)
    #pragma unroll
    for (int j = 0; j < 8; ++j)
      u.h[g*8 + j] = p[g*16 + j];
  return u.v;
}

// B fragment (32x16 f16) pre-packed lane-major: one 32B load per lane.
__device__ __forceinline__ v16h load_b_frag(const h16* frag, int ln) {
  return *reinterpret_cast<const v16h*>(frag + ln * 16);
}

// rcp-based activations: v_exp_f32 + v_rcp_f32, no precise-div chains
__device__ __forceinline__ float sigm(float x) {
  return __builtin_amdgcn_rcpf(1.f + __expf(-x));
}
__device__ __forceinline__ float tanh_fast(float x) {
  const float ax = fabsf(x);
  const float e  = __expf(-2.f * ax);            // (0,1]
  const float t  = 1.f - 2.f * e * __builtin_amdgcn_rcpf(1.f + e);
  return copysignf(t, x);
}
__device__ __forceinline__ float softplus_f(float x) {
  return fmaxf(x, 0.f) + log1pf(__expf(-fabsf(x)));
}

// ---------------------------------------------------------------- prep kernels

// Pack W (ncols x K, row-major f32) into WMMA-B f16 fragments.
__global__ void pack_w(const float* __restrict__ W, h16* __restrict__ dst,
                       int K, int ktiles, int total) {
  int e = blockIdx.x * blockDim.x + threadIdx.x;
  if (e >= total) return;
  int i    = e & 15;
  int ln   = (e >> 4) & 31;
  int frag = e >> 9;
  int kt   = frag % ktiles;
  int nt   = frag / ktiles;
  int col  = nt * 16 + (ln & 15);
  int k    = kt * 32 + ((i >> 3) << 4) + ((ln >> 4) << 3) + (i & 7);
  dst[(size_t)frag * 512 + ln * 16 + i] = (h16)W[(size_t)col * K + k];
}

__global__ void h2half(const float* __restrict__ src, h16* __restrict__ dst, int n) {
  int t = blockIdx.x * blockDim.x + threadIdx.x;
  if (t < n) dst[t] = (h16)src[t];
}

// per-row: first-occurrence index (argmax of equality) + flipped walk rows
__global__ void prep_rows(const int* __restrict__ walks,
                          int* __restrict__ uidx, int* __restrict__ wrow) {
  int b = blockIdx.x * blockDim.x + threadIdx.x;
  if (b >= BB) return;
  int w[LL];
  #pragma unroll
  for (int i = 0; i < LL; ++i) w[i] = walks[(size_t)b * LL + i];
  int un[LL];
  #pragma unroll
  for (int i = 0; i < LL; ++i) {
    int u = i;
    #pragma unroll
    for (int j = LL - 1; j >= 0; --j) if (w[j] == w[i]) u = j;  // smallest match
    un[i] = u;
  }
  #pragma unroll
  for (int t = 0; t < LL; ++t) {
    uidx[(size_t)b * LL + t] = un[LL - 1 - t];
    wrow[(size_t)b * LL + t] = w[LL - 1 - t];
  }
}

__global__ void zero_misc(float* __restrict__ acc, int* __restrict__ flag) {
  int t = blockIdx.x * blockDim.x + threadIdx.x;
  if (t < 4) acc[t] = 0.f;
  int f = t - 4;
  if (f >= 0 && f < NN) flag[f] = 0;
}

__global__ void count_idxs(const int* __restrict__ idxs, int* __restrict__ flag) {
  int t = threadIdx.x;
  if (t < SUBN) atomicAdd(&flag[idxs[t]], 1);
}

// Sum of y_t (for pos_w = count / sum)
__global__ void sum_yt(const int* __restrict__ walks, const float* __restrict__ y0,
                       const int* __restrict__ idxs, float* __restrict__ acc) {
  int t = blockIdx.x * blockDim.x + threadIdx.x;
  if (t >= LOSS_COUNT) return;
  int ccol = t & 63;
  int r = t >> 6;
  int tau = r % (LL - 1); r /= (LL - 1);
  int j = r % SUBN;
  int s = r / SUBN;
  int node = idxs[j];
  int wf = walks[((size_t)s * NN + node) * LL + (LL - 2 - tau)];  // walks_f[tau+1]
  atomicAdd(&acc[2], y0[(size_t)wf * ORIG + ccol]);
}

__global__ void finalize(const float* __restrict__ acc, float* __restrict__ out_loss) {
  const float count = (float)LOSS_COUNT;
  const float pos_w = count / acc[2];
  out_loss[0] = (pos_w * acc[0] + acc[1]) / count;
}

// ---------------------------------------------------------------- fused GRU

__global__ __launch_bounds__(32)
void rum_gru_kernel(const h16*  __restrict__ hh,
                    const float* __restrict__ Wih_w, const float* __restrict__ bih_w,
                    const float* __restrict__ bhh_w,
                    const float* __restrict__ bih,   const float* __restrict__ bhh,
                    const h16*  __restrict__ pWhhW,
                    const h16*  __restrict__ pWhh,
                    const h16*  __restrict__ pWih,
                    const int*  __restrict__ g_uidx, const int* __restrict__ g_wrow,
                    const int*  __restrict__ flag,
                    const float* __restrict__ y0,  const float* __restrict__ W_ss,
                    const float* __restrict__ b_ss,
                    float* __restrict__ out, float* __restrict__ acc)
{
  __shared__ float sh_state[16 * HD];        // fp32 main-GRU state (single buffer)
  __shared__ h16   sh_hf16[2][16 * HD];      // f16 shadow state (A source, main)
  __shared__ h16   sh_ywalk[LL * 16 * HD];   // walk-GRU outputs (= walk states)
  __shared__ h16   sh_hw[16 * HD];           // gathered h rows for current step
  __shared__ h16   sh_giw[LL * 384];         // gi lookup table: [u][col]
  __shared__ float sh_wb[4 * 128];           // walk bias combos: bR,bZ,bI,bN
  __shared__ float sh_mb[4 * 128];           // main bias combos
  __shared__ int   sh_uidx[16 * LL];
  __shared__ int   sh_wrow[16 * LL];

  const int ln = threadIdx.x;
  const int b0 = blockIdx.x * 16;
  const int c  = ln & 15;

  // ---- stage per-block constants -----------------------------------------
  if (ln < 16) {
    #pragma unroll
    for (int t = 0; t < LL; ++t) {
      sh_uidx[ln * LL + t] = g_uidx[(size_t)(b0 + ln) * LL + t];
      sh_wrow[ln * LL + t] = g_wrow[(size_t)(b0 + ln) * LL + t];
    }
  }
  for (int i = ln; i < LL * 384; i += 32) {          // gi table (transposed Wih_w)
    int col = i % 384, u = i / 384;
    sh_giw[i] = (h16)Wih_w[col * LL + u];
  }
  for (int i = ln; i < 128; i += 32) {               // bias combos
    sh_wb[i]       = bih_w[i]       + bhh_w[i];
    sh_wb[128 + i] = bih_w[128 + i] + bhh_w[128 + i];
    sh_wb[256 + i] = bih_w[256 + i];
    sh_wb[384 + i] = bhh_w[256 + i];
    sh_mb[i]       = bih[i]       + bhh[i];
    sh_mb[128 + i] = bih[128 + i] + bhh[128 + i];
    sh_mb[256 + i] = bih[256 + i];
    sh_mb[384 + i] = bhh[256 + i];
  }

  // warm L2/WGP$ for the packed weights (global_prefetch_b8)
  __builtin_prefetch(pWhhW + (ln << 6), 0, 1);
  __builtin_prefetch(pWhh  + (ln << 6), 0, 1);
  __builtin_prefetch(pWih  + (ln << 6), 0, 1);

  // ---------------- walk GRU ----------------------------------------------
  // state after step ts IS sh_ywalk[ts]; h0 = 0 -> ts==0 has no matmul.
  for (int ts = 0; ts < LL; ++ts) {
    const h16* prev = sh_ywalk + (size_t)(ts - 1) * 16 * HD;   // valid for ts>0
    for (int t = 0; t < 8; ++t) {
      const int colr = t * 16 + c;
      v8f Fr, Fz, Fin, Fhn;
      const float bR = sh_wb[colr];
      const float bZ = sh_wb[128 + colr];
      const float bI = sh_wb[256 + colr];
      const float bN = sh_wb[384 + colr];
      #pragma unroll
      for (int r = 0; r < 8; ++r) {
        const int m = ((ln >> 4) << 3) + r;
        const int u = sh_uidx[m * LL + ts];
        Fr[r]  = (float)sh_giw[u * 384 + colr]       + bR;
        Fz[r]  = (float)sh_giw[u * 384 + 128 + colr] + bZ;
        Fin[r] = (float)sh_giw[u * 384 + 256 + colr] + bI;
        Fhn[r] = bN;
      }
      if (ts) {
        #pragma unroll
        for (int kt = 0; kt < 4; ++kt) {
          v16h A  = load_a_half(prev, ln, kt * 32);
          v16h Br = load_b_frag(pWhhW + ((size_t)(t     ) * 4 + kt) * 512, ln);
          v16h Bz = load_b_frag(pWhhW + ((size_t)(t +  8) * 4 + kt) * 512, ln);
          v16h Bn = load_b_frag(pWhhW + ((size_t)(t + 16) * 4 + kt) * 512, ln);
          Fr  = wmma16(A, Br, Fr);
          Fz  = wmma16(A, Bz, Fz);
          Fhn = wmma16(A, Bn, Fhn);
        }
      }
      #pragma unroll
      for (int r = 0; r < 8; ++r) {
        const int m = ((ln >> 4) << 3) + r;
        const float hp = ts ? (float)prev[m * HD + colr] : 0.f;
        const float rr = sigm(Fr[r]);
        const float zz = sigm(Fz[r]);
        const float nn = tanh_fast(Fin[r] + rr * Fhn[r]);
        const float hn = (1.f - zz) * nn + zz * hp;
        sh_ywalk[(ts * 16 + m) * HD + colr] = (h16)hn;
      }
    }
  }

  // main-GRU h0 = h_walk = sh_ywalk[7]; seed the fp32 state from it.
  for (int i = ln; i < 16 * HD; i += 32)
    sh_state[i] = (float)sh_ywalk[(LL - 1) * 16 * HD + i];

  // ---------------- main GRU (x = [h[walk] | y_walk], 256-wide) ------------
  float sa = 0.f, sb = 0.f;
  const h16* asrc = sh_ywalk + (size_t)(LL - 1) * 16 * HD;   // f16 A-state source
  for (int ts = 0; ts < LL; ++ts) {
    { // gather this step's h rows into LDS (f16), 2 lanes per row
      const int m   = ln >> 1;
      const int off = (ln & 1) * 64;
      const int node = sh_wrow[m * LL + ts];
      const uint4* src = reinterpret_cast<const uint4*>(hh + (size_t)node * FF + off);
      uint4* dst = reinterpret_cast<uint4*>(sh_hw + m * FF + off);
      #pragma unroll
      for (int j = 0; j < 8; ++j) dst[j] = src[j];
    }
    h16* wbuf = sh_hf16[ts & 1];
    const h16* yw = sh_ywalk + (size_t)ts * 16 * HD;
    for (int t = 0; t < 8; ++t) {
      const int colr = t * 16 + c;
      v8f Fr, Fz, Fin, Fhn;
      const float bR = sh_mb[colr];
      const float bZ = sh_mb[128 + colr];
      const float bI = sh_mb[256 + colr];
      const float bN = sh_mb[384 + colr];
      #pragma unroll
      for (int r = 0; r < 8; ++r) { Fr[r] = bR; Fz[r] = bZ; Fin[r] = bI; Fhn[r] = bN; }
      #pragma unroll
      for (int kt = 0; kt < 4; ++kt) {   // h . Whh^T
        v16h A  = load_a_half(asrc, ln, kt * 32);
        v16h Br = load_b_frag(pWhh + ((size_t)(t     ) * 4 + kt) * 512, ln);
        v16h Bz = load_b_frag(pWhh + ((size_t)(t +  8) * 4 + kt) * 512, ln);
        v16h Bn = load_b_frag(pWhh + ((size_t)(t + 16) * 4 + kt) * 512, ln);
        Fr  = wmma16(A, Br, Fr);
        Fz  = wmma16(A, Bz, Fz);
        Fhn = wmma16(A, Bn, Fhn);
      }
      #pragma unroll
      for (int kt = 0; kt < 8; ++kt) {   // x . Wih^T  (K = 256)
        v16h A = (kt < 4) ? load_a_half(sh_hw, ln, kt * 32)
                          : load_a_half(yw,    ln, (kt - 4) * 32);
        v16h Br = load_b_frag(pWih + ((size_t)(t     ) * 8 + kt) * 512, ln);
        v16h Bz = load_b_frag(pWih + ((size_t)(t +  8) * 8 + kt) * 512, ln);
        v16h Bn = load_b_frag(pWih + ((size_t)(t + 16) * 8 + kt) * 512, ln);
        Fr  = wmma16(A, Br, Fr);
        Fz  = wmma16(A, Bz, Fz);
        Fin = wmma16(A, Bn, Fin);
      }
      #pragma unroll
      for (int r = 0; r < 8; ++r) {
        const int m = ((ln >> 4) << 3) + r;
        const float hp = sh_state[m * HD + colr];
        const float rr = sigm(Fr[r]);
        const float zz = sigm(Fz[r]);
        const float nn = tanh_fast(Fin[r] + rr * Fhn[r]);
        const float hn = (1.f - zz) * nn + zz * hp;
        sh_state[m * HD + colr] = hn;       // own-column RMW, no cross-chunk dep
        wbuf[m * HD + colr]     = (h16)hn;  // f16 shadow for next step's A
      }
    }
    asrc = wbuf;

    // ------- self-supervision loss partials (rare flagged rows) -------
    if (ts < LL - 1) {
      for (int m = 0; m < 16; ++m) {
        const int b  = b0 + m;
        const int s_ = b / NN;
        const int n_ = b - s_ * NN;
        const int cnt = flag[n_];
        if (cnt > 0) {
          const int nextnode = sh_wrow[m * LL + ts + 1];
          #pragma unroll
          for (int cc2 = 0; cc2 < 2; ++cc2) {
            const int cc = ln * 2 + cc2;
            float d = b_ss[cc];
            for (int k = 0; k < FF; ++k) d += sh_state[m * FF + k] * W_ss[(size_t)cc * FF + k];
            const float yt = y0[(size_t)nextnode * ORIG + cc];
            sa += (float)cnt * yt * softplus_f(-d);
            sb += (float)cnt * (1.f - yt) * softplus_f(d);
          }
        }
      }
    }
  }

  if (sa != 0.f || sb != 0.f) {
    atomicAdd(&acc[0], sa);
    atomicAdd(&acc[1], sb);
  }

  // write hT (fp32), 2 lanes per row
  {
    const int m   = ln >> 1;
    const int off = (ln & 1) * 64;
    #pragma unroll
    for (int j = 0; j < 16; ++j)
      reinterpret_cast<float4*>(out + (size_t)(b0 + m) * HD + off)[j] =
          reinterpret_cast<const float4*>(sh_state + m * HD + off)[j];
  }
}

// ---------------------------------------------------------------- launch

extern "C" void kernel_launch(void* const* d_in, const int* in_sizes, int n_in,
                              void* d_out, int out_size, void* d_ws, size_t ws_size,
                              hipStream_t stream) {
  (void)in_sizes; (void)n_in; (void)out_size; (void)ws_size;
  const float* h_f   = (const float*)d_in[0];
  const float* y0    = (const float*)d_in[1];
  const float* Wih_w = (const float*)d_in[2];
  const float* Whh_w = (const float*)d_in[3];
  const float* bih_w = (const float*)d_in[4];
  const float* bhh_w = (const float*)d_in[5];
  const float* Wih   = (const float*)d_in[6];
  const float* Whh   = (const float*)d_in[7];
  const float* bih   = (const float*)d_in[8];
  const float* bhh   = (const float*)d_in[9];
  const float* W_ss  = (const float*)d_in[10];
  const float* b_ss  = (const float*)d_in[11];
  const int*   walks = (const int*)d_in[12];
  const int*   idxs  = (const int*)d_in[13];
  float* out = (float*)d_out;

  char* wsb = (char*)d_ws;
  size_t off = 0;
  auto alloc = [&](size_t bytes) -> void* {
    void* p = wsb + off;
    off = (off + bytes + 255) & ~size_t(255);
    return p;
  };
  float* acc   = (float*)alloc(4 * sizeof(float));           // SA, SB, Syt
  int*   flag  = (int*)  alloc((size_t)NN * sizeof(int));
  h16*   hh    = (h16*)  alloc((size_t)NN * FF * sizeof(h16));
  h16*   pWhhW = (h16*)  alloc((size_t)24 * 4 * 512 * sizeof(h16));
  h16*   pWhh  = (h16*)  alloc((size_t)24 * 4 * 512 * sizeof(h16));
  h16*   pWih  = (h16*)  alloc((size_t)24 * 8 * 512 * sizeof(h16));
  int*   uidx  = (int*)  alloc((size_t)BB * LL * sizeof(int));
  int*   wrow  = (int*)  alloc((size_t)BB * LL * sizeof(int));

  zero_misc<<<(NN + 4 + 255) / 256, 256, 0, stream>>>(acc, flag);
  count_idxs<<<1, 128, 0, stream>>>(idxs, flag);
  pack_w<<<(384 * 128 + 255) / 256, 256, 0, stream>>>(Whh_w, pWhhW, 128, 4, 384 * 128);
  pack_w<<<(384 * 128 + 255) / 256, 256, 0, stream>>>(Whh,   pWhh,  128, 4, 384 * 128);
  pack_w<<<(384 * 256 + 255) / 256, 256, 0, stream>>>(Wih,   pWih,  256, 8, 384 * 256);
  h2half<<<(NN * FF + 255) / 256, 256, 0, stream>>>(h_f, hh, NN * FF);
  prep_rows<<<(BB + 127) / 128, 128, 0, stream>>>(walks, uidx, wrow);
  sum_yt<<<(LOSS_COUNT + 255) / 256, 256, 0, stream>>>(walks, y0, idxs, acc);

  rum_gru_kernel<<<BB / 16, 32, 0, stream>>>(hh, Wih_w, bih_w, bhh_w, bih, bhh,
                                             pWhhW, pWhh, pWih, uidx, wrow, flag,
                                             y0, W_ss, b_ss, out, acc);

  finalize<<<1, 1, 0, stream>>>(acc, out + (size_t)SS * NN * HD);
}
